// hRNN_69741678953277
// MI455X (gfx1250) — matrix operand order repaired and checked by reading, not compiled, not run on hardware
//
#include <hip/hip_runtime.h>
#include <hip/hip_bf16.h>

#define BATCH 16
#define TT    4096
#define NU    64
#define NX    256
#define NY    32
#define LDA   (NX + 8)   // padded LDS row stride (halves): 528B = 33*16B, shifts 4 banks/row

typedef __attribute__((ext_vector_type(16))) _Float16 v16h;
typedef __attribute__((ext_vector_type(8)))  _Float16 v8h;
typedef __attribute__((ext_vector_type(8)))  float    v8f;

union V16HU { v16h v; v8h h[2]; };

__device__ __forceinline__ v8f wmma16x16x32(v16h a, v16h b, v8f c) {
  // D = A(16x32 f16) * B(32x16 f16) + C(16x16 f32)
  return __builtin_amdgcn_wmma_f32_16x16x32_f16(false, a, false, b, (short)0, c,
                                                false, false);
}

// ---- fragment loaders -------------------------------------------------------
// A fragment 16x32 f16, contiguous k order as in the B loaders: lane m=lane&15, group g=lane>>4;
// halves 0..7  -> K = kbase + g*16 + 0..7
// halves 8..15 -> K = kbase + g*16 + 8 + 0..7
__device__ __forceinline__ v16h load_a_f32(const float* abase, int ld, int kbase,
                                           int lane) {
  int m = lane & 15, g = (lane >> 4) & 1;
  const float* p = abase + (size_t)m * ld + kbase + g * 16;
  v16h a;
#pragma unroll
  for (int i = 0; i < 8; ++i) a[i] = (_Float16)p[i];
#pragma unroll
  for (int i = 0; i < 8; ++i) a[8 + i] = (_Float16)p[8 + i];
  return a;
}

// A fragment from padded fp16 LDS buffer [16][LDA]: two conflict-free ds_load_b128
__device__ __forceinline__ v16h load_a_lds16(const _Float16* buf, int kc, int lane) {
  int m = lane & 15, g = (lane >> 4) & 1;
  const _Float16* p = buf + m * LDA + kc * 32 + g * 16;
  V16HU u;
  u.h[0] = *(const v8h*)(p);
  u.h[1] = *(const v8h*)(p + 8);
  return u.v;
}

// B-matrix 32x16 f16 layout: lane n=n0+(lane&15), group g=lane>>4;
// half j -> K = kbase + g*16 + j.
// Case 1: B[k][n] = W[n][k]  (W row-major [N][K], k contiguous)
__device__ __forceinline__ v16h load_bT_f32(const float* W, int ldw, int n0,
                                            int kbase, int lane) {
  int n = n0 + (lane & 15), g = (lane >> 4) & 1;
  const float* p = W + (size_t)n * ldw + kbase + g * 16;
  v16h b;
#pragma unroll
  for (int i = 0; i < 16; ++i) b[i] = (_Float16)p[i];
  return b;
}

// Case 2: B[k][n] = M[k][n]  (M row-major [K][N], n contiguous, k strided)
__device__ __forceinline__ v16h load_b_f32(const float* Bm, int ldb, int n0,
                                           int kbase, int lane) {
  int n = n0 + (lane & 15), g = (lane >> 4) & 1;
  const float* p = Bm + (size_t)(kbase + g * 16) * ldb + n;
  v16h b;
#pragma unroll
  for (int i = 0; i < 16; ++i) b[i] = (_Float16)p[(size_t)i * ldb];
  return b;
}

// ---- kernel 1: Ku[b,t,:] = inputs[b,t,:] @ K_w^T, stored time-shifted at
// states slot t+1 (slot t holds the Ku value consumed when computing h_t) ----
__global__ __launch_bounds__(128) void ku_kernel(const float* __restrict__ inputs,
                                                 const float* __restrict__ Kw,
                                                 float* __restrict__ states) {
  int wid  = blockIdx.x * 4 + (threadIdx.x >> 5);
  int lane = threadIdx.x & 31;
  int nt = wid & 15;        // 16 n-tiles over NX
  int mt = wid >> 4;        // 4096 m-tiles over B*T
  int b  = mt >> 8;         // T/16 = 256 tiles per batch
  int t0 = (mt & 255) << 4;
  const float* abase = inputs + ((size_t)b * TT + t0) * NU;

  v8f acc = {};
#pragma unroll
  for (int kc = 0; kc < 2; ++kc) {
    v16h a  = load_a_f32(abase, NU, kc * 32, lane);
    v16h bm = load_bT_f32(Kw, NU, nt * 16, kc * 32, lane);  // B[k][n]=Kw[n][k]
    acc = wmma16x16x32(a, bm, acc);
  }

  int g = lane >> 4;
  int col = nt * 16 + (lane & 15);
#pragma unroll
  for (int r = 0; r < 8; ++r) {
    int t = t0 + r + 8 * g;
    if (t + 1 < TT)
      states[(size_t)b * TT * NX + (size_t)(t + 1) * NX + col] = acc[r];
  }
}

// ---- kernel 2: persistent single-workgroup scan (8 waves x 2 n-tiles) -------
__global__ __launch_bounds__(256) void scan_kernel(const float* __restrict__ h0,
                                                   const float* __restrict__ Hw,
                                                   const float* __restrict__ Hb,
                                                   const float* __restrict__ Einv,
                                                   float* states) {
  __shared__ _Float16 hbuf[BATCH * LDA];   // current state, fp16, padded rows
  __shared__ _Float16 xbuf[BATCH * LDA];   // relu(x), fp16, padded rows

  int lane = threadIdx.x & 31;
  int w    = threadIdx.x >> 5;   // wave id 0..7, owns n-tiles 2w and 2w+1
  int g    = lane >> 4;
  int nl   = lane & 15;
  int n0   = w * 32 + nl;        // column of tile 2w
  int n1   = n0 + 16;            // column of tile 2w+1

  // Recurrent weights resident in VGPRs: 32 x v16h = 256 VGPRs.
  v16h B1[2][8], B2[2][8];
#pragma unroll
  for (int kc = 0; kc < 8; ++kc) {
    B1[0][kc] = load_bT_f32(Hw, NX, w * 32,      kc * 32, lane);  // h @ H_w^T
    B1[1][kc] = load_bT_f32(Hw, NX, w * 32 + 16, kc * 32, lane);
    B2[0][kc] = load_b_f32(Einv, NX, w * 32,      kc * 32, lane); // relu_x @ E_inv
    B2[1][kc] = load_b_f32(Einv, NX, w * 32 + 16, kc * 32, lane);
  }
  float hb0 = Hb[n0], hb1 = Hb[n1];

  // h <- h0 (LDS fp16, padded) and states slot 0 <- h0 (fp32)
#pragma unroll
  for (int i = 0; i < 16; ++i) {
    int idx = threadIdx.x * 16 + i;      // 256*16 == BATCH*NX == 4096
    float v = h0[idx];
    int b = idx >> 8, x = idx & 255;
    hbuf[b * LDA + x] = (_Float16)v;
    states[(size_t)b * TT * NX + x] = v;
  }
  __syncthreads();

  // Software-pipelined ku gather: slot t holds Ku[t-1] until overwritten by h_t.
  float ku0[8], ku1[8];
  {
    const float* kup = states + (size_t)1 * NX;
#pragma unroll
    for (int r = 0; r < 8; ++r) {
      size_t rowoff = (size_t)(r + 8 * g) * TT * NX;
      ku0[r] = kup[rowoff + n0];
      ku1[r] = kup[rowoff + n1];
    }
  }

  for (int t = 1; t < TT; ++t) {
    // Prefetch next step's ku (slot t+1 is untouched during iteration t).
    // Final iteration prefetches a clamped dummy slot (discarded).
    int tn = (t + 1 < TT) ? (t + 1) : t;
    float nk0[8], nk1[8];
    {
      const float* kup = states + (size_t)tn * NX;
#pragma unroll
      for (int r = 0; r < 8; ++r) {
        size_t rowoff = (size_t)(r + 8 * g) * TT * NX;
        nk0[r] = kup[rowoff + n0];
        nk1[r] = kup[rowoff + n1];
      }
    }

    // x = h @ H_w^T : two independent accumulator chains share each A fragment
    v8f a0 = {}, a1 = {};
#pragma unroll
    for (int kc = 0; kc < 8; ++kc) {
      v16h a = load_a_lds16(hbuf, kc, lane);
      a0 = wmma16x16x32(a, B1[0][kc], a0);
      a1 = wmma16x16x32(a, B1[1][kc], a1);
    }

    // x += H_b + ku ; relu ; publish fp16
#pragma unroll
    for (int r = 0; r < 8; ++r) {
      int row = r + 8 * g;
      float v0 = a0[r] + hb0 + ku0[r]; v0 = v0 > 0.0f ? v0 : 0.0f;
      float v1 = a1[r] + hb1 + ku1[r]; v1 = v1 > 0.0f ? v1 : 0.0f;
      xbuf[row * LDA + n0] = (_Float16)v0;
      xbuf[row * LDA + n1] = (_Float16)v1;
    }
    __syncthreads();

    // h = relu_x @ E_inv
    v8f c0 = {}, c1 = {};
#pragma unroll
    for (int kc = 0; kc < 8; ++kc) {
      v16h a = load_a_lds16(xbuf, kc, lane);
      c0 = wmma16x16x32(a, B2[0][kc], c0);
      c1 = wmma16x16x32(a, B2[1][kc], c1);
    }

#pragma unroll
    for (int r = 0; r < 8; ++r) {
      int row = r + 8 * g;
      float v0 = c0[r], v1 = c1[r];
      hbuf[row * LDA + n0] = (_Float16)v0;
      hbuf[row * LDA + n1] = (_Float16)v1;
      size_t go = (size_t)row * TT * NX + (size_t)t * NX;
      states[go + n0] = v0;
      states[go + n1] = v1;
    }
    __syncthreads();

    // rotate pipeline registers
#pragma unroll
    for (int r = 0; r < 8; ++r) { ku0[r] = nk0[r]; ku1[r] = nk1[r]; }
  }
}

// ---- kernel 3: yest = states @ out_w^T + out_b ------------------------------
__global__ __launch_bounds__(128) void out_kernel(const float* __restrict__ states,
                                                  const float* __restrict__ Ow,
                                                  const float* __restrict__ Ob,
                                                  float* __restrict__ yest) {
  int wid  = blockIdx.x * 4 + (threadIdx.x >> 5);
  int lane = threadIdx.x & 31;
  int nt = wid & 1;          // 2 n-tiles over NY=32
  int mt = wid >> 1;         // 4096 m-tiles over B*T

  const float* abase = states + (size_t)mt * 16 * NX;
  v8f acc = {};
#pragma unroll
  for (int kc = 0; kc < 8; ++kc) {
    v16h a  = load_a_f32(abase, NX, kc * 32, lane);
    v16h bm = load_bT_f32(Ow, NX, nt * 16, kc * 32, lane);  // B[k][n]=Ow[n][k]
    acc = wmma16x16x32(a, bm, acc);
  }

  int g = lane >> 4;
  int col = nt * 16 + (lane & 15);
  float ob = Ob[col];
#pragma unroll
  for (int r = 0; r < 8; ++r) {
    size_t row = (size_t)mt * 16 + r + 8 * g;
    yest[row * NY + col] = acc[r] + ob;
  }
}

extern "C" void kernel_launch(void* const* d_in, const int* in_sizes, int n_in,
                              void* d_out, int out_size, void* d_ws, size_t ws_size,
                              hipStream_t stream) {
  const float* inputs = (const float*)d_in[0];  // [B,T,NU]
  const float* h0     = (const float*)d_in[1];  // [B,NX]
  const float* Hw     = (const float*)d_in[2];  // [NX,NX]
  const float* Hb     = (const float*)d_in[3];  // [NX]
  const float* Kw     = (const float*)d_in[4];  // [NX,NU]
  const float* Einv   = (const float*)d_in[5];  // [NX,NX]
  const float* Ow     = (const float*)d_in[6];  // [NY,NX]
  const float* Ob     = (const float*)d_in[7];  // [NY]
  (void)in_sizes; (void)n_in; (void)d_ws; (void)ws_size; (void)out_size;

  float* yest   = (float*)d_out;                      // [B,T,NY]
  float* states = yest + (size_t)BATCH * TT * NY;     // [B,T,NX]

  // 1) Ku GEMM (time-shifted into the states region)
  ku_kernel<<<(4096 * 16) / 4, 128, 0, stream>>>(inputs, Kw, states);
  // 2) serial scan: one persistent workgroup, weights in VGPRs
  scan_kernel<<<1, 256, 0, stream>>>(h0, Hw, Hb, Einv, states);
  // 3) output projection
  out_kernel<<<(4096 * 2) / 4, 128, 0, stream>>>(states, Ow, Ob, yest);
}
